// Attention_163208757296
// MI455X (gfx1250) — matrix operand order repaired and checked by reading, not compiled, not run on hardware
//
#include <hip/hip_runtime.h>

// ---------------------------------------------------------------------------
// Causal multi-head attention forward for MI455X (gfx1250, wave32, WMMA).
// All GEMMs run on v_wmma_f32_16x16x32_f16 with f32 accumulation.
// Prep kernels convert x -> f16 and weights -> f16 transposed once, so every
// steady-state fragment load reads f16 directly (A: two 16B loads/lane, B: strided gather) and the
// GEMM inner loops contain no f32->f16 conversion VALU at all.
// ---------------------------------------------------------------------------

typedef __attribute__((ext_vector_type(16))) _Float16 v16h;
typedef __attribute__((ext_vector_type(8)))  _Float16 v8h;
typedef __attribute__((ext_vector_type(4)))  _Float16 v4h;
typedef __attribute__((ext_vector_type(8)))  float    v8f;

#define BATCH   4
#define NHEAD   16
#define DHEAD   64
#define NCTX    2048
#define DMODEL  1024
#define NROWS   (BATCH * NCTX)        // 8192 rows of x / z

// --------------------------- WMMA wrapper ----------------------------------
__device__ __forceinline__ v8f wmma_f16(v16h a, v16h b, v8f c) {
  // (neg_a, A, neg_b, B, c_mod, C, reuse_a, reuse_b)
  return __builtin_amdgcn_wmma_f32_16x16x32_f16(false, a, false, b,
                                                (short)0, c, false, false);
}

// --------------------------- Fragment loaders ------------------------------
// A-matrix 16x32 f16 (ISA 7.12.2): lane(0..15)=row M, elems 0..7 = K[half*8..],
// elems 8..15 = K[16+half*8..]; lanes 16..31 same rows, K offset +8/+24.
__device__ __forceinline__ v16h a_frag_f16(const _Float16* rowptr, int kbase, int half) {
  v8h g0 = *(const v8h*)(rowptr + kbase + half * 8);
  v8h g1 = *(const v8h*)(rowptr + kbase + 16 + half * 8);
  return __builtin_shufflevector(g0, g1, 0, 1, 2, 3, 4, 5, 6, 7,
                                         8, 9, 10, 11, 12, 13, 14, 15);
}

// B-matrix 32x16 f16: lane%16 = column N, elems 0..7 = K[half*8..],
// elems 8..15 = K[16+half*8..] (same K order as A); strided gather, row stride ld.
__device__ __forceinline__ v16h b_frag_f16c(const _Float16* p, size_t ld, int half) {
  v16h v;
#pragma unroll
  for (int s = 0; s < 16; ++s) {
    const int kk = (s < 8) ? (half * 8 + s) : (16 + half * 8 + (s - 8));
    v[s] = p[(size_t)kk * ld];
  }
  return v;
}

// ---------------------------------------------------------------------------
// Kernel 0a: x prep.  x (f32, [8192][1024]) -> Xh (f16, same layout).
// One float4 load + one 8B f16 store per thread; pure streaming pass.
// ---------------------------------------------------------------------------
__global__ __launch_bounds__(256) void xprep_kernel(
    const float* __restrict__ x, _Float16* __restrict__ Xh) {
  const size_t idx = ((size_t)blockIdx.x * 256 + threadIdx.x) * 4;
  const float4 v = *(const float4*)(x + idx);
  v4h h;
  h[0] = (_Float16)v.x; h[1] = (_Float16)v.y;
  h[2] = (_Float16)v.z; h[3] = (_Float16)v.w;
  *(v4h*)(Xh + idx) = h;
}

// ---------------------------------------------------------------------------
// Kernel 0b: weight prep.  W[n][k] (f32) -> Wt[k][n] (f16), 1024x1024 each.
// 32x32 LDS tile transpose, coalesced on both sides.  mats: Wq,Wk,Wv,Wo.
// ---------------------------------------------------------------------------
__global__ __launch_bounds__(256) void wprep_kernel(
    const float* __restrict__ Wq, const float* __restrict__ Wk,
    const float* __restrict__ Wv, const float* __restrict__ Wo,
    _Float16* __restrict__ Wq_t, _Float16* __restrict__ Wk_t,
    _Float16* __restrict__ Wv_t, _Float16* __restrict__ Wo_t) {
  __shared__ float tile[32][33];
  const int tiles_per_mat = (DMODEL / 32) * (DMODEL / 32);   // 1024
  const int flat = blockIdx.x;
  const int mat  = flat / tiles_per_mat;
  const int rem  = flat % tiles_per_mat;
  const int tr   = (rem >> 5) * 32;     // n-block
  const int tc   = (rem & 31) * 32;     // k-block
  const float* W = (mat == 0) ? Wq : (mat == 1) ? Wk : (mat == 2) ? Wv : Wo;
  _Float16* Wt   = (mat == 0) ? Wq_t : (mat == 1) ? Wk_t : (mat == 2) ? Wv_t : Wo_t;

  const int tx = threadIdx.x & 31;      // along k (read) / n (write)
  const int ty = threadIdx.x >> 5;      // 0..7
#pragma unroll
  for (int jj = 0; jj < 4; ++jj) {
    const int n = tr + ty + jj * 8;
    tile[ty + jj * 8][tx] = W[(size_t)n * DMODEL + tc + tx];
  }
  __syncthreads();
#pragma unroll
  for (int jj = 0; jj < 4; ++jj) {
    const int k = tc + ty + jj * 8;
    Wt[(size_t)k * DMODEL + tr + tx] = (_Float16)tile[tx][ty + jj * 8];
  }
}

// ---------------------------------------------------------------------------
// Kernel 1: fused QKV projection.  Y[row, nout] = sum_k Xh[row,k] * Wt[k,nout]
// One wave -> 32x64 output tile (2 A-frags x 4 B-frags -> 8 WMMA / k-step).
// Q:[b,h,q,dh]  K^T:[b,h,dh,p]  V:[b,h,p,dh], all f16.
// ---------------------------------------------------------------------------
__global__ __launch_bounds__(256) void qkv_kernel(
    const _Float16* __restrict__ Xh,
    const _Float16* __restrict__ Wq_t, const _Float16* __restrict__ Wk_t,
    const _Float16* __restrict__ Wv_t,
    _Float16* __restrict__ Qb, _Float16* __restrict__ Kt,
    _Float16* __restrict__ Vb) {
  const int lane = threadIdx.x & 31;
  const int wave = threadIdx.x >> 5;
  const int half = lane >> 4;
  const int r    = lane & 15;

  const int row_tiles = NROWS / 32;                 // 256
  const int flat = blockIdx.x * 8 + wave;
  const int mat  = flat / (row_tiles * 16);
  const int rem  = flat % (row_tiles * 16);
  const int rowb = (rem >> 4) * 32;
  const int colb = (rem & 15) * 64;

  const _Float16* Wt = (mat == 0) ? Wq_t : (mat == 1) ? Wk_t : Wv_t;
  const _Float16* arow0 = Xh + (size_t)(rowb + r) * DMODEL;
  const _Float16* arow1 = Xh + (size_t)(rowb + 16 + r) * DMODEL;

  v8f c[2][4] = {};
  for (int k = 0; k < DMODEL; k += 32) {
    const v16h a0 = a_frag_f16(arow0, k, half);
    const v16h a1 = a_frag_f16(arow1, k, half);
    const _Float16* wrow = Wt + (size_t)k * DMODEL + colb + r;
#pragma unroll
    for (int t = 0; t < 4; ++t) {
      const v16h b = b_frag_f16c(wrow + t * 16, DMODEL, half);
      c[0][t] = wmma_f16(a0, b, c[0][t]);
      c[1][t] = wmma_f16(a1, b, c[1][t]);
    }
  }

  // C/D layout: lane<16 -> (M=j, N=lane); lane>=16 -> (M=8+j, N=lane-16)
#pragma unroll
  for (int rb = 0; rb < 2; ++rb) {
    const int row0 = rowb + rb * 16 + half * 8;
#pragma unroll
    for (int t = 0; t < 4; ++t) {
      const int col = colb + t * 16 + r;
      const int h   = col >> 6;
      const int dh  = col & 63;
#pragma unroll
      for (int j = 0; j < 8; ++j) {
        const int row = row0 + j;
        const int b_  = row >> 11;            // /NCTX
        const int q   = row & (NCTX - 1);
        const size_t bh = (size_t)b_ * NHEAD + h;
        const _Float16 v = (_Float16)c[rb][t][j];
        if (mat == 0)      Qb[(bh * NCTX + q) * DHEAD + dh] = v;
        else if (mat == 1) Kt[(bh * DHEAD + dh) * NCTX + q] = v;
        else               Vb[(bh * NCTX + q) * DHEAD + dh] = v;
      }
    }
  }
}

// ---------------------------------------------------------------------------
// Kernel 2: causal flash attention.  One wave per (b,h,16-q-row tile).
// Online softmax; P converted C-layout -> A-frag layout via per-wave LDS tile.
// Reference masks with -1e10 BEFORE the 1/sqrt(64) scale => masked = -1.25e9.
// ---------------------------------------------------------------------------
__global__ __launch_bounds__(256) void attn_kernel(
    const _Float16* __restrict__ Qb, const _Float16* __restrict__ Kt,
    const _Float16* __restrict__ Vb, _Float16* __restrict__ Zb) {
  __shared__ _Float16 lds[8][16 * 32];     // one 16x32 f16 P-tile per wave

  const int lane = threadIdx.x & 31;
  const int wave = threadIdx.x >> 5;
  const int half = lane >> 4;
  const int r    = lane & 15;

  const int flat  = blockIdx.x * 8 + wave;
  const int qt    = flat & 127;            // 128 q tiles per (b,h)
  const int bh    = flat >> 7;
  const int qbase = qt * 16;

  const _Float16* Qp = Qb + ((size_t)bh * NCTX + qbase) * DHEAD;
  const _Float16* Kp = Kt + (size_t)bh * DHEAD * NCTX;   // [dh][p]
  const _Float16* Vp = Vb + (size_t)bh * NCTX * DHEAD;   // [p][dh]

  const v16h qa0 = a_frag_f16(Qp + (size_t)r * DHEAD, 0, half);
  const v16h qa1 = a_frag_f16(Qp + (size_t)r * DHEAD, 32, half);

  float m[8], l[8];
#pragma unroll
  for (int j = 0; j < 8; ++j) { m[j] = -3.0e38f; l[j] = 0.0f; }
  v8f o[4] = {};

  _Float16* myl = &lds[wave][0];
  const int qend = qbase + 15;

  for (int pb = 0; pb <= qend; pb += 32) {
    // ---- scores S = Q . K^T  (two 16x16 tiles: cols pb..+15, pb+16..+31)
    v8f s0 = {}, s1 = {};
    {
      v16h kb = b_frag_f16c(Kp + (size_t)0 * NCTX + pb + r, NCTX, half);
      s0 = wmma_f16(qa0, kb, s0);
      kb = b_frag_f16c(Kp + (size_t)32 * NCTX + pb + r, NCTX, half);
      s0 = wmma_f16(qa1, kb, s0);
      kb = b_frag_f16c(Kp + (size_t)0 * NCTX + pb + 16 + r, NCTX, half);
      s1 = wmma_f16(qa0, kb, s1);
      kb = b_frag_f16c(Kp + (size_t)32 * NCTX + pb + 16 + r, NCTX, half);
      s1 = wmma_f16(qa1, kb, s1);
    }
    // speculative prefetch of next K/V blocks (global_prefetch_b8)
    __builtin_prefetch(Kp + (size_t)lane * NCTX + pb + 32, 0, 0);
    __builtin_prefetch(Vp + (size_t)(pb + 32 + lane) * DHEAD, 0, 0);

    // ---- scale + causal mask (C layout: row = qbase+half*8+j, col = pb(+16)+r)
    float a0[8], a1[8], mx[8];
#pragma unroll
    for (int j = 0; j < 8; ++j) {
      const int row = qbase + half * 8 + j;
      a0[j] = (pb + r      <= row) ? s0[j] * 0.125f : -1.25e9f;
      a1[j] = (pb + 16 + r <= row) ? s1[j] * 0.125f : -1.25e9f;
      mx[j] = fmaxf(a0[j], a1[j]);
    }
    // row max: reduce over the 16-lane half holding this row
#pragma unroll
    for (int j = 0; j < 8; ++j) {
#pragma unroll
      for (int off = 1; off < 16; off <<= 1)
        mx[j] = fmaxf(mx[j], __shfl_xor(mx[j], off, 32));
    }

    float alpha[8], rs[8];
#pragma unroll
    for (int j = 0; j < 8; ++j) {
      const float mn = fmaxf(m[j], mx[j]);
      alpha[j] = __expf(m[j] - mn);
      m[j]     = mn;
      a0[j]    = __expf(a0[j] - mn);
      a1[j]    = __expf(a1[j] - mn);
      rs[j]    = a0[j] + a1[j];
    }
#pragma unroll
    for (int j = 0; j < 8; ++j) {
#pragma unroll
      for (int off = 1; off < 16; off <<= 1)
        rs[j] += __shfl_xor(rs[j], off, 32);
    }
#pragma unroll
    for (int j = 0; j < 8; ++j) l[j] = l[j] * alpha[j] + rs[j];
#pragma unroll
    for (int t = 0; t < 4; ++t) {
#pragma unroll
      for (int j = 0; j < 8; ++j) o[t][j] *= alpha[j];
    }

    // ---- P (C layout) -> LDS -> A-fragment layout
#pragma unroll
    for (int j = 0; j < 8; ++j) {
      myl[(half * 8 + j) * 32 + r]      = (_Float16)a0[j];
      myl[(half * 8 + j) * 32 + 16 + r] = (_Float16)a1[j];
    }
    asm volatile("s_wait_dscnt 0" ::: "memory");   // LDS RAW across lanes
    const v16h pa = a_frag_f16(myl + r * 32, 0, half);

    // ---- O += P . V   (B rows = key positions, N = head dims, contiguous)
#pragma unroll
    for (int t = 0; t < 4; ++t) {
      const v16h vb = b_frag_f16c(Vp + (size_t)pb * DHEAD + t * 16 + r, DHEAD, half);
      o[t] = wmma_f16(pa, vb, o[t]);
    }
  }

  // ---- normalize and write z_flat [b, q, h*64+dh] as f16
  const int b_ = bh >> 4;
  const int h  = bh & 15;
#pragma unroll
  for (int t = 0; t < 4; ++t) {
#pragma unroll
    for (int j = 0; j < 8; ++j) {
      const int q = qbase + half * 8 + j;
      const float v = o[t][j] / l[j];
      Zb[((size_t)b_ * NCTX + q) * DMODEL + h * DHEAD + t * 16 + r] = (_Float16)v;
    }
  }
}

// ---------------------------------------------------------------------------
// Kernel 3: output projection  out[row, d] = sum_f z[row,f] * Wo_t[f,d]
// One wave -> 32x64 output tile.
// ---------------------------------------------------------------------------
__global__ __launch_bounds__(256) void proj_kernel(
    const _Float16* __restrict__ Zb, const _Float16* __restrict__ Wo_t,
    float* __restrict__ out) {
  const int lane = threadIdx.x & 31;
  const int wave = threadIdx.x >> 5;
  const int half = lane >> 4;
  const int r    = lane & 15;

  const int flat = blockIdx.x * 8 + wave;
  const int rowb = (flat >> 4) * 32;
  const int colb = (flat & 15) * 64;

  const _Float16* arow0 = Zb + (size_t)(rowb + r) * DMODEL;
  const _Float16* arow1 = Zb + (size_t)(rowb + 16 + r) * DMODEL;

  v8f c[2][4] = {};
  for (int k = 0; k < DMODEL; k += 32) {
    const v16h a0 = a_frag_f16(arow0, k, half);
    const v16h a1 = a_frag_f16(arow1, k, half);
    const _Float16* wrow = Wo_t + (size_t)k * DMODEL + colb + r;
#pragma unroll
    for (int t = 0; t < 4; ++t) {
      const v16h b = b_frag_f16c(wrow + t * 16, DMODEL, half);
      c[0][t] = wmma_f16(a0, b, c[0][t]);
      c[1][t] = wmma_f16(a1, b, c[1][t]);
    }
  }

#pragma unroll
  for (int rb = 0; rb < 2; ++rb) {
#pragma unroll
    for (int t = 0; t < 4; ++t) {
#pragma unroll
      for (int j = 0; j < 8; ++j) {
        out[(size_t)(rowb + rb * 16 + half * 8 + j) * DMODEL +
            colb + t * 16 + r] = c[rb][t][j];
      }
    }
  }
}

// ---------------------------------------------------------------------------
extern "C" void kernel_launch(void* const* d_in, const int* in_sizes, int n_in,
                              void* d_out, int out_size, void* d_ws, size_t ws_size,
                              hipStream_t stream) {
  (void)in_sizes; (void)n_in; (void)out_size; (void)ws_size;

  // setup_inputs() dict order: x, W_K, W_Q, W_V, W_O
  const float* x  = (const float*)d_in[0];
  const float* Wk = (const float*)d_in[1];
  const float* Wq = (const float*)d_in[2];
  const float* Wv = (const float*)d_in[3];
  const float* Wo = (const float*)d_in[4];
  float* out = (float*)d_out;

  // Workspace layout (f16):
  //   Qb/Kt/Vb/Zb: 4 x B*H*N*DH = 8,388,608 elems (16 MiB each)
  //   Xh: NROWS*DMODEL = 8,388,608 elems (16 MiB)
  //   Wq_t/Wk_t/Wv_t/Wo_t: 4 x 1024*1024 elems (2 MiB each)
  const size_t elems  = (size_t)BATCH * NHEAD * NCTX * DHEAD;
  const size_t welems = (size_t)DMODEL * DMODEL;
  _Float16* Qb   = (_Float16*)d_ws;
  _Float16* Kt   = Qb + elems;        // K transposed: [b,h,dh,p]
  _Float16* Vb   = Kt + elems;        // V:            [b,h,p,dh]
  _Float16* Zb   = Vb + elems;        // z_flat f16:   [b*q, 1024]
  _Float16* Xh   = Zb + elems;        // x as f16:     [b*q, 1024]
  _Float16* Wq_t = Xh + elems;
  _Float16* Wk_t = Wq_t + welems;
  _Float16* Wv_t = Wk_t + welems;
  _Float16* Wo_t = Wv_t + welems;

  // x prep: 8,388,608 elems / 4 per thread / 256 per block = 8192 blocks
  xprep_kernel<<<dim3(8192), dim3(256), 0, stream>>>(x, Xh);
  // weight prep: 4 mats * (1024/32)^2 tiles = 4096 blocks
  wprep_kernel<<<dim3(4096), dim3(256), 0, stream>>>(Wq, Wk, Wv, Wo,
                                                     Wq_t, Wk_t, Wv_t, Wo_t);
  // 256 row-tiles * 16 col-tiles * 3 matrices = 12288 waves / 8 per block
  qkv_kernel<<<dim3(1536), dim3(256), 0, stream>>>(Xh, Wq_t, Wk_t, Wv_t,
                                                   Qb, Kt, Vb);
  // 4*16*128 = 8192 waves / 8 per block
  attn_kernel<<<dim3(1024), dim3(256), 0, stream>>>(Qb, Kt, Vb, Zb);
  // 256*16 = 4096 waves / 8 per block
  proj_kernel<<<dim3(512), dim3(256), 0, stream>>>(Zb, Wo_t, out);
}
